// Seq2SeqDynamic_29454885716594
// MI455X (gfx1250) — matrix operand order repaired and checked by reading, not compiled, not run on hardware
//
#include <hip/hip_runtime.h>
#include <hip/hip_bf16.h>

// ---------------------------------------------------------------------------
// MI455X (gfx1250) seq2seq: bi-GRU encoder + Bahdanau attention decoder.
// bf16 WMMA (v_wmma_f32_16x16x32_bf16, fp32 accum) for all GEMMs; batched
// input projections; factored attention energy; single batched output-head
// GEMM. GEMM K-loop: rolled, triple-buffered (mod-3), with the issue order
// pinned via __builtin_amdgcn_sched_group_barrier (2 WMMA : 5 VMEM-read) so
// refill loads overlap WMMAs instead of being sunk onto their consumers.
// ---------------------------------------------------------------------------

typedef __attribute__((ext_vector_type(16))) __bf16 v16bf;
typedef __attribute__((ext_vector_type(8)))  float  v8f;

__device__ __forceinline__ unsigned short f2bf(float f) {
  unsigned int u = __builtin_bit_cast(unsigned int, f);
  return (unsigned short)((u + 0x7FFFu + ((u >> 16) & 1u)) >> 16);  // RNE
}

__device__ __forceinline__ float sigmoidf_(float x) {
  return 1.0f / (1.0f + __expf(-x));
}

struct Q2 { uint4 a, b; };

__device__ __forceinline__ v16bf load_frag2(const unsigned short* p0,
                                            const unsigned short* p1) {
  Q2 q;
  q.a = *reinterpret_cast<const uint4*>(p0);
  q.b = *reinterpret_cast<const uint4*>(p1);
  return __builtin_bit_cast(v16bf, q);
}

// ---------------------------------------------------------------------------
// C[M,N] = A[M,K] * W[N,K]^T + bias[N]   (A,W bf16 row-major, C fp32)
// One wave -> 16 x 64 output strip (A fragment reused over 4 WMMAs).
// K compile-time, K % 32 == 0, K >= 192 (call sites: 512/1024/1536/2048).
// ---------------------------------------------------------------------------

// Consume chunk held in buffer BUF (4 WMMAs); refill buffer BUF from element
// offset KOFF. BUF must be a compile-time constant expression.
#define GEMM_CONSUME(BUF)                                                     \
  _Pragma("unroll")                                                           \
  for (int i = 0; i < NT; ++i)                                                \
    acc[i] = __builtin_amdgcn_wmma_f32_16x16x32_bf16(                         \
        false, a[(BUF)], false, b[(BUF)][i], (short)0, acc[i], false, false);
#define GEMM_REFILL(BUF, KOFF)                                                \
  a[(BUF)] = loadA(KOFF);                                                     \
  _Pragma("unroll")                                                           \
  for (int i = 0; i < NT; ++i) b[(BUF)][i] = loadB(i, (KOFF));

template <int K>
__global__ __launch_bounds__(128)
void wmma_gemm_bf16(const unsigned short* __restrict__ A,
                    const unsigned short* __restrict__ W,
                    const float* __restrict__ bias,
                    float* __restrict__ C,
                    int M, int N) {
  const int NT = 4;
  constexpr int NCH   = K / 32;          // 32-deep K chunks
  constexpr int R2    = NCH % 3;         // leftover chunks-with-refill
  constexpr int NTRIP = (NCH - 3 - R2) / 3;
  constexpr int CB    = 3 * NTRIP;       // first chunk after the main loop

  int lane = threadIdx.x & 31;
  int wave = threadIdx.x >> 5;
  int g = lane >> 4;        // K-half group
  int m = lane & 15;        // row within tile / col within C tile
  long ngroups = N / (16 * NT);
  long mtiles  = M / 16;
  long task = (long)blockIdx.x * 4 + wave;
  if (task >= mtiles * ngroups) return;
  long tm  = task / ngroups;
  long tn0 = (task % ngroups) * NT;   // first N-tile index

  // Fold lane-group offset into base pointers: loads are base + imm offset.
  const unsigned short* ArowG = A + (tm * 16 + m) * (long)K + g * 8;
  const unsigned short* BrowG[NT];
#pragma unroll
  for (int i = 0; i < NT; ++i)
    BrowG[i] = W + ((tn0 + i) * 16 + m) * (long)K + g * 8;

  // Fragment loaders per CDNA5 ISA 7.12.2 (16-bit A 16x32 / B 32x16 layouts).
  auto loadA = [&](int k0) {
    return load_frag2(ArowG + k0, ArowG + k0 + 16);
  };
  auto loadB = [&](int i, int k0) {
    return load_frag2(BrowG[i] + k0, BrowG[i] + k0 + 16);
  };

  v8f zero = {};
  v8f acc[NT];
#pragma unroll
  for (int i = 0; i < NT; ++i) acc[i] = zero;

  // ---- prologue: preload chunks 0..2 into the three buffer sets ----
  v16bf a[3];
  v16bf b[3][NT];
#pragma unroll
  for (int c = 0; c < 3; ++c) {
    a[c] = loadA(c * 32);
#pragma unroll
    for (int i = 0; i < NT; ++i) b[c][i] = loadB(i, c * 32);
  }

  // ---- main loop: consume chunks kc..kc+2 (bufs 0,1,2), refill +3 each.
  // sched_group_barrier pins "2 WMMA : 5 VMEM-read" so the first WMMA group
  // waits only on last iteration's loads (partial s_wait_loadcnt) and the
  // 30 refill loads issue interleaved with the 12 WMMAs.
  int kc = 0;
  for (int j = 0; j < NTRIP; ++j, kc += 96) {
    GEMM_CONSUME(0); GEMM_REFILL(0, kc + 96);
    GEMM_CONSUME(1); GEMM_REFILL(1, kc + 128);
    GEMM_CONSUME(2); GEMM_REFILL(2, kc + 160);
    __builtin_prefetch(BrowG[0] + kc + 2048, 0, 1);  // global_prefetch_b8, L2 warm
#pragma unroll
    for (int u = 0; u < 6; ++u) {
      __builtin_amdgcn_sched_group_barrier(0x008, 2, 0);  // 2 x WMMA
      __builtin_amdgcn_sched_group_barrier(0x020, 5, 0);  // 5 x VMEM read
    }
  }

  // ---- compile-time epilogue: R2 chunks with refill, then last 3 dry ----
  if (R2 >= 1) { GEMM_CONSUME(CB % 3);       GEMM_REFILL(CB % 3, (CB + 3) * 32); }
  if (R2 >= 2) { GEMM_CONSUME((CB + 1) % 3); GEMM_REFILL((CB + 1) % 3, (CB + 4) * 32); }
  GEMM_CONSUME((NCH - 3) % 3);
  GEMM_CONSUME((NCH - 2) % 3);
  GEMM_CONSUME((NCH - 1) % 3);

#pragma unroll
  for (int i = 0; i < NT; ++i) {
    long col = (tn0 + i) * 16 + m;
    float bv = bias ? bias[col] : 0.0f;
#pragma unroll
    for (int j = 0; j < 8; ++j) {
      long row = tm * 16 + j + 8 * g;     // C layout: VGPR j -> M=j(+8 for hi lanes)
      C[row * (long)N + col] = acc[i][j] + bv;
    }
  }
}

#undef GEMM_CONSUME
#undef GEMM_REFILL

template <int K>
static void gemm_launch(const unsigned short* A, const unsigned short* W,
                        const float* bias, float* C, int M, int N,
                        hipStream_t st) {
  long tasks = (long)(M / 16) * (N / 64);
  wmma_gemm_bf16<K><<<(int)((tasks + 3) / 4), 128, 0, st>>>(A, W, bias, C, M, N);
}

// ---------------------------------------------------------------------------
// fp32 -> bf16 weight conversion, optionally column-strided (for Wa split)
// ---------------------------------------------------------------------------
__global__ __launch_bounds__(256)
void convert_strided_kernel(const float* __restrict__ src,
                            unsigned short* __restrict__ dst,
                            long rows, long cols, long src_ld, long src_off) {
  long i = (long)blockIdx.x * 256 + threadIdx.x;
  if (i >= rows * cols) return;
  long r = i / cols, c = i - r * cols;
  dst[i] = f2bf(src[r * src_ld + src_off + c]);
}

__global__ __launch_bounds__(256)
void fill_zero_kernel(unsigned int* __restrict__ p, long n) {
  long i = (long)blockIdx.x * 256 + threadIdx.x;
  if (i < n) p[i] = 0u;
}

// ---------------------------------------------------------------------------
// Embedding gathers (fp32 table -> bf16 activations)
// ---------------------------------------------------------------------------
__global__ __launch_bounds__(256)
void embed_gather_kernel(const int* __restrict__ tok, const float* __restrict__ emb,
                         unsigned short* __restrict__ out, long rows, int E) {
  long i = (long)blockIdx.x * 256 + threadIdx.x;
  if (i >= rows * (long)E) return;
  long r = i / E; int c = (int)(i - r * E);
  out[i] = f2bf(emb[(size_t)tok[r] * E + c]);
}

__global__ __launch_bounds__(256)
void embed_dec_kernel(const int* __restrict__ src, const int* __restrict__ trg,
                      const float* __restrict__ emb, unsigned short* __restrict__ out,
                      int T, int Bn, int S, int E) {
  long i = (long)blockIdx.x * 256 + threadIdx.x;
  if (i >= (long)T * Bn * E) return;
  long r = i / E; int c = (int)(i - r * E);
  int t = (int)(r / Bn), b = (int)(r - (long)t * Bn);
  int tokid = (t == 0) ? src[(S - 1) * Bn + b] : trg[(t - 1) * Bn + b];
  out[i] = f2bf(emb[(size_t)tokid * E + c]);
}

// ---------------------------------------------------------------------------
// Fused GRU gates: h = (1-z)*n + z*h ; biases already folded into gi/gh GEMMs.
// ---------------------------------------------------------------------------
__global__ __launch_bounds__(256)
void gru_gate_kernel(const float* __restrict__ gi, const float* __restrict__ gh,
                     float* __restrict__ h, unsigned short* __restrict__ hbf,
                     float* __restrict__ out_copy, unsigned short* __restrict__ hc,
                     int Bn, int H) {
  int i = blockIdx.x * 256 + threadIdx.x;
  if (i >= Bn * H) return;
  int b = i / H, k = i - b * H;
  const float* gib = gi + (size_t)b * 3 * H;
  const float* ghb = gh + (size_t)b * 3 * H;
  float r = sigmoidf_(gib[k] + ghb[k]);
  float z = sigmoidf_(gib[H + k] + ghb[H + k]);
  float n = tanhf(gib[2 * H + k] + r * ghb[2 * H + k]);
  float hv = (1.0f - z) * n + z * h[i];
  h[i] = hv;
  hbf[i] = f2bf(hv);
  if (out_copy) out_copy[i] = hv;
  if (hc) hc[(size_t)b * 2 * H + k] = f2bf(hv);
}

// enc_bt[b,s,:] = out_f[s,b,:] + out_b[S-1-s,b,:]  (+ bf16 copy for GEMM A)
__global__ __launch_bounds__(256)
void enc_combine_kernel(const float* __restrict__ outf, const float* __restrict__ outb,
                        float* __restrict__ encbt, unsigned short* __restrict__ encbt_bf,
                        int S, int Bn, int H) {
  long i = (long)blockIdx.x * 256 + threadIdx.x;
  if (i >= (long)S * Bn * H) return;
  int k = (int)(i % H); long r = i / H; int s = (int)(r % S); int b = (int)(r / S);
  float v = outf[((long)s * Bn + b) * H + k] + outb[((long)(S - 1 - s) * Bn + b) * H + k];
  encbt[i] = v;
  encbt_bf[i] = f2bf(v);
}

// scores[b,s] = sum_k relu(hWa[b,k] + encproj[b,s,k]) * v[k]   (wave per (b,s))
__global__ __launch_bounds__(256)
void attn_score_kernel(const float* __restrict__ hWa, const float* __restrict__ encproj,
                       const float* __restrict__ vvec, float* __restrict__ scores,
                       int Bn, int S, int H) {
  int wave = (blockIdx.x * 256 + threadIdx.x) >> 5;
  int lane = threadIdx.x & 31;
  if (wave >= Bn * S) return;
  int b = wave / S, s = wave - b * S;
  const float* hp = hWa + (size_t)b * H;
  const float* ep = encproj + ((size_t)b * S + s) * H;
  float sum = 0.0f;
  for (int k = lane; k < H; k += 32) {
    float e = hp[k] + ep[k];
    e = e > 0.0f ? e : 0.0f;
    sum += e * vvec[k];
  }
  for (int o = 16; o > 0; o >>= 1) sum += __shfl_down(sum, o, 32);  // wave32
  if (lane == 0) scores[wave] = sum;
}

// copy e_t into rnn_in[:, 0:E]
__global__ __launch_bounds__(256)
void copy_rnnin_emb_kernel(const unsigned short* __restrict__ src,
                           unsigned short* __restrict__ rnn, int Bn, int E, int H) {
  int i = blockIdx.x * 256 + threadIdx.x;
  if (i >= Bn * E) return;
  int b = i / E, c = i - b * E;
  rnn[(size_t)b * (E + H) + c] = src[i];
}

// softmax over S + context; writes bf16 ctx into rnn_in[:,E:] and hc[:,H:2H]
__global__ __launch_bounds__(256)
void softmax_ctx_kernel(const float* __restrict__ scores, const float* __restrict__ encbt,
                        unsigned short* __restrict__ rnn_ctx,
                        unsigned short* __restrict__ hc_ctx,
                        int S, int H, int E) {
  int b = blockIdx.x;
  __shared__ float a[64];
  if (threadIdx.x < S) a[threadIdx.x] = scores[b * S + threadIdx.x];
  __syncthreads();
  if (threadIdx.x == 0) {
    float mx = a[0];
    for (int s = 1; s < S; ++s) mx = fmaxf(mx, a[s]);
    float sum = 0.0f;
    for (int s = 0; s < S; ++s) { a[s] = __expf(a[s] - mx); sum += a[s]; }
    float inv = 1.0f / sum;
    for (int s = 0; s < S; ++s) a[s] *= inv;
  }
  __syncthreads();
  for (int k = threadIdx.x; k < H; k += 256) {
    const float* ep = encbt + ((size_t)b * S) * H + k;
    float acc = 0.0f;
    for (int s = 0; s < S; ++s) acc += a[s] * ep[(size_t)s * H];
    unsigned short bv = f2bf(acc);
    rnn_ctx[(size_t)b * (E + H) + k] = bv;
    hc_ctx[(size_t)b * (2 * H) + k] = bv;
  }
}

// ---------------------------------------------------------------------------
extern "C" void kernel_launch(void* const* d_in, const int* in_sizes, int n_in,
                              void* d_out, int out_size, void* d_ws, size_t ws_size,
                              hipStream_t stream) {
  (void)in_sizes; (void)n_in; (void)out_size; (void)ws_size;
  const int S = 64, T = 64, Bn = 32, E = 512, H = 1024, V = 32000;
  const int H3 = 3 * H, H2 = 2 * H, EH = E + H;

  const int*   src   = (const int*)d_in[0];
  const int*   trg   = (const int*)d_in[1];
  const float* emb   = (const float*)d_in[2];
  const float* Wih_f = (const float*)d_in[3];
  const float* Whh_f = (const float*)d_in[4];
  const float* bih_f = (const float*)d_in[5];
  const float* bhh_f = (const float*)d_in[6];
  const float* Wih_b = (const float*)d_in[7];
  const float* Whh_b = (const float*)d_in[8];
  const float* bih_b = (const float*)d_in[9];
  const float* bhh_b = (const float*)d_in[10];
  const float* Wa    = (const float*)d_in[11];
  const float* ba    = (const float*)d_in[12];
  const float* vvec  = (const float*)d_in[13];
  const float* Wih_d = (const float*)d_in[14];
  const float* Whh_d = (const float*)d_in[15];
  const float* bih_d = (const float*)d_in[16];
  const float* bhh_d = (const float*)d_in[17];
  const float* Wout  = (const float*)d_in[18];
  const float* bout  = (const float*)d_in[19];

  size_t off = 0;
  auto alloc = [&](size_t bytes) -> void* {
    off = (off + 255) & ~(size_t)255;
    void* p = (char*)d_ws + off;
    off += bytes;
    return p;
  };

  // bf16 weights
  unsigned short* wihf_b = (unsigned short*)alloc((size_t)H3 * E * 2);
  unsigned short* whhf_b = (unsigned short*)alloc((size_t)H3 * H * 2);
  unsigned short* wihb_b = (unsigned short*)alloc((size_t)H3 * E * 2);
  unsigned short* whhb_b = (unsigned short*)alloc((size_t)H3 * H * 2);
  unsigned short* wah_b  = (unsigned short*)alloc((size_t)H * H * 2);
  unsigned short* wae_b  = (unsigned short*)alloc((size_t)H * H * 2);
  unsigned short* wihd_b = (unsigned short*)alloc((size_t)H3 * EH * 2);
  unsigned short* whhd_b = (unsigned short*)alloc((size_t)H3 * H * 2);
  unsigned short* wout_b = (unsigned short*)alloc((size_t)V * H2 * 2);
  // activations
  unsigned short* semb   = (unsigned short*)alloc((size_t)S * Bn * E * 2);
  unsigned short* demb   = (unsigned short*)alloc((size_t)T * Bn * E * 2);
  float* gif     = (float*)alloc((size_t)S * Bn * H3 * 4);
  float* gib_    = (float*)alloc((size_t)S * Bn * H3 * 4);
  float* outf    = (float*)alloc((size_t)S * Bn * H * 4);
  float* outb    = (float*)alloc((size_t)S * Bn * H * 4);
  float* encbt   = (float*)alloc((size_t)Bn * S * H * 4);
  unsigned short* encbt_b = (unsigned short*)alloc((size_t)Bn * S * H * 2);
  float* encproj = (float*)alloc((size_t)Bn * S * H * 4);
  // h states: one contiguous zeroable block [hf f32][hb f32][hfb bf16][hbb bf16]
  float* hf = (float*)alloc((size_t)Bn * H * 12);
  float* hb = hf + (size_t)Bn * H;
  unsigned short* hfb = (unsigned short*)(hb + (size_t)Bn * H);
  unsigned short* hbb = hfb + (size_t)Bn * H;
  float* ghf   = (float*)alloc((size_t)Bn * H3 * 4);
  float* ghb   = (float*)alloc((size_t)Bn * H3 * 4);
  float* hWa   = (float*)alloc((size_t)Bn * H * 4);
  float* scores = (float*)alloc((size_t)Bn * S * 4);
  unsigned short* rnn_in = (unsigned short*)alloc((size_t)Bn * EH * 2);
  float* gid   = (float*)alloc((size_t)Bn * H3 * 4);
  float* ghd   = (float*)alloc((size_t)Bn * H3 * 4);
  unsigned short* hc_all = (unsigned short*)alloc((size_t)T * Bn * H2 * 2);

  auto conv = [&](const float* s, unsigned short* d, long rows, long cols,
                  long ld, long co) {
    long n = rows * cols;
    convert_strided_kernel<<<(int)((n + 255) / 256), 256, 0, stream>>>(s, d, rows, cols, ld, co);
  };

  // ---- weight conversion (fp32 -> bf16) ----
  conv(Wih_f, wihf_b, H3, E, E, 0);
  conv(Whh_f, whhf_b, H3, H, H, 0);
  conv(Wih_b, wihb_b, H3, E, E, 0);
  conv(Whh_b, whhb_b, H3, H, H, 0);
  conv(Wa, wah_b, H, H, H2, 0);    // Wa[:, :H]
  conv(Wa, wae_b, H, H, H2, H);    // Wa[:, H:]
  conv(Wih_d, wihd_b, H3, EH, EH, 0);
  conv(Whh_d, whhd_b, H3, H, H, 0);
  conv(Wout, wout_b, V, H2, H2, 0);

  // ---- embeddings ----
  embed_gather_kernel<<<((long)S * Bn * E + 255) / 256, 256, 0, stream>>>(
      src, emb, semb, (long)S * Bn, E);
  embed_dec_kernel<<<((long)T * Bn * E + 255) / 256, 256, 0, stream>>>(
      src, trg, emb, demb, T, Bn, S, E);

  // ---- batched encoder input projections (hoisted out of the recurrence) ----
  gemm_launch<512>(semb, wihf_b, bih_f, gif, S * Bn, H3, stream);
  gemm_launch<512>(semb, wihb_b, bih_b, gib_, S * Bn, H3, stream);

  // ---- zero h states ----
  fill_zero_kernel<<<(int)(((size_t)Bn * H * 3 + 255) / 256), 256, 0, stream>>>(
      (unsigned int*)hf, (long)Bn * H * 3);

  // ---- encoder recurrence (fwd + bwd) ----
  for (int t = 0; t < S; ++t) {
    gemm_launch<1024>(hfb, whhf_b, bhh_f, ghf, Bn, H3, stream);
    gru_gate_kernel<<<128, 256, 0, stream>>>(gif + (size_t)t * Bn * H3, ghf, hf, hfb,
                                             outf + (size_t)t * Bn * H, nullptr, Bn, H);
    gemm_launch<1024>(hbb, whhb_b, bhh_b, ghb, Bn, H3, stream);
    gru_gate_kernel<<<128, 256, 0, stream>>>(gib_ + (size_t)(S - 1 - t) * Bn * H3, ghb,
                                             hb, hbb, outb + (size_t)t * Bn * H,
                                             nullptr, Bn, H);
  }

  // ---- combine directions; precompute step-invariant Wa_e * enc + ba ----
  enc_combine_kernel<<<(int)(((long)Bn * S * H + 255) / 256), 256, 0, stream>>>(
      outf, outb, encbt, encbt_b, S, Bn, H);
  gemm_launch<1024>(encbt_b, wae_b, ba, encproj, Bn * S, H, stream);

  // ---- decoder (hf/hfb continue from final fwd hidden = h_fwd) ----
  for (int t = 0; t < T; ++t) {
    gemm_launch<1024>(hfb, wah_b, nullptr, hWa, Bn, H, stream);
    attn_score_kernel<<<(Bn * S * 32 + 255) / 256, 256, 0, stream>>>(
        hWa, encproj, vvec, scores, Bn, S, H);
    copy_rnnin_emb_kernel<<<(Bn * E + 255) / 256, 256, 0, stream>>>(
        demb + (size_t)t * Bn * E, rnn_in, Bn, E, H);
    softmax_ctx_kernel<<<Bn, 256, 0, stream>>>(
        scores, encbt, rnn_in + E, hc_all + (size_t)t * Bn * H2 + H, S, H, E);
    gemm_launch<1536>(rnn_in, wihd_b, bih_d, gid, Bn, H3, stream);
    gemm_launch<1024>(hfb, whhd_b, bhh_d, ghd, Bn, H3, stream);
    gru_gate_kernel<<<128, 256, 0, stream>>>(gid, ghd, hf, hfb, nullptr,
                                             hc_all + (size_t)t * Bn * H2, Bn, H);
  }

  // ---- single batched output head: [T*B,2H] x [2H,V] (Wout read once) ----
  gemm_launch<2048>(hc_all, wout_b, bout, (float*)d_out, T * Bn, V, stream);
}
